// MaskedMultiHeadSelfAttention_64020782514316
// MI455X (gfx1250) — matrix operand order repaired and checked by reading, not compiled, not run on hardware
//
#include <hip/hip_runtime.h>
#include <math.h>

// ---------------------------------------------------------------------------
// Causal multi-head self-attention, MI455X (gfx1250, wave32, WMMA).
// Heavy math: v_wmma_f32_16x16x32_bf16 (fp32 accumulate).
// V staging:  global_load_async_to_lds_b128 (ASYNCcnt) overlapped with QK^T.
// GEMMs:      ping-pong double buffering (no inter-buffer register copies).
// ---------------------------------------------------------------------------

typedef unsigned short u16;
typedef unsigned int   u32;
typedef unsigned long long u64;
typedef __attribute__((ext_vector_type(16))) __bf16        v16bf;
typedef __attribute__((ext_vector_type(8)))  float         v8f;
typedef __attribute__((ext_vector_type(8)))  unsigned int  u32x8;
typedef __attribute__((ext_vector_type(16))) unsigned short u16x16;

static constexpr int Bn = 2;
static constexpr int Tn = 2048;
static constexpr int Cn = 1024;
static constexpr int Hn = 16;
static constexpr int Dn = 64;
static constexpr int Mn = Bn * Tn;      // 4096 rows of x

// fp32 -> bf16 round-to-nearest-even
__device__ __forceinline__ u16 f2bf(float f) {
    u32 u = __builtin_bit_cast(u32, f);
    u32 r = u + 0x7FFFu + ((u >> 16) & 1u);
    return (u16)(r >> 16);
}

// butterfly reductions across the 16 lanes of a half-wave (xor masks < 16
// keep lanes 0-15 and 16-31 independent on wave32)
__device__ __forceinline__ float half16_max(float v) {
    #pragma unroll
    for (int o = 1; o < 16; o <<= 1) v = fmaxf(v, __shfl_xor(v, o, 32));
    return v;
}
__device__ __forceinline__ float half16_sum(float v) {
    #pragma unroll
    for (int o = 1; o < 16; o <<= 1) v += __shfl_xor(v, o, 32);
    return v;
}

// A-operand K offset for packed-pair p (contiguous order: k = 16*lh + 2p, matches B)
__device__ __forceinline__ int a_koff(int p, int lh) {
    int kq = 2 * p;
    return kq + (lh ? 16 : 0);
}

// ---------------------------------------------------------------------------
// Conversion kernels
// ---------------------------------------------------------------------------
__global__ void k_f32_to_bf16(const float* __restrict__ in, u16* __restrict__ out, int n) {
    int i = blockIdx.x * blockDim.x + threadIdx.x;
    if (i < n) out[i] = f2bf(in[i]);
}

// Wt[n*C + k] = bf16(W[k*C + n])  (1024x1024 fixed)
__global__ void k_w_transpose_bf16(const float* __restrict__ W, u16* __restrict__ Wt) {
    int i = blockIdx.x * blockDim.x + threadIdx.x;   // 0 .. 1M-1
    int n = i >> 10;
    int k = i & 1023;
    Wt[i] = f2bf(W[k * Cn + n]);
}

// ---------------------------------------------------------------------------
// GEMM fragment loader: A = 16x32 slice of a row-major [*,C] matrix,
// B = 4 x (32x16) slices of the transposed weight (row of Wt = column of W).
// ---------------------------------------------------------------------------
__device__ __forceinline__ void load_step(const u16* __restrict__ arow,
                                          const u16* __restrict__ wt_n,
                                          int k0, int lh, int nc,
                                          u32x8& a, u32x8 b[4]) {
    #pragma unroll
    for (int p = 0; p < 8; ++p)
        a[p] = *(const u32*)(arow + k0 + a_koff(p, lh));
    #pragma unroll
    for (int nt = 0; nt < 4; ++nt) {
        const u16* bcol = wt_n + (size_t)(nt * 16 + nc) * Cn + k0 + lh * 16;
        #pragma unroll
        for (int p = 0; p < 8; ++p) b[nt][p] = *(const u32*)(bcol + 2 * p);
    }
}

__device__ __forceinline__ void wmma4(const u32x8& a8, const u32x8 b8[4], v8f acc[4]) {
    v16bf a = __builtin_bit_cast(v16bf, a8);
    #pragma unroll
    for (int nt = 0; nt < 4; ++nt) {
        v16bf b = __builtin_bit_cast(v16bf, b8[nt]);
        acc[nt] = __builtin_amdgcn_wmma_f32_16x16x32_bf16(
            false, a, false, b, (short)0, acc[nt], false, false);
    }
}

// Ping-pong double-buffered 16(M) x 64(N) x C accumulation: buffers alternate
// roles every 32-wide k-step, so no register rotation copies are needed.
__device__ __forceinline__ void gemm_16x64(const u16* __restrict__ arow,
                                           const u16* __restrict__ wt_n,
                                           int lh, int nc, v8f acc[4]) {
    u32x8 aA, bA[4], aB, bB[4];
    load_step(arow, wt_n, 0, lh, nc, aA, bA);
    int k0 = 0;
    for (; k0 + 64 < Cn; k0 += 64) {
        load_step(arow, wt_n, k0 + 32, lh, nc, aB, bB);   // prefetch odd step
        wmma4(aA, bA, acc);                               // consume even step
        load_step(arow, wt_n, k0 + 64, lh, nc, aA, bA);   // prefetch even step
        wmma4(aB, bB, acc);                               // consume odd step
    }
    load_step(arow, wt_n, k0 + 32, lh, nc, aB, bB);       // last odd step
    wmma4(aA, bA, acc);
    wmma4(aB, bB, acc);
}

// ---------------------------------------------------------------------------
// QKV projection: out[b,h,t,d] = (x @ W + bias) * scale, bf16 output.
// ---------------------------------------------------------------------------
__global__ void k_gemm_qkv(const u16* __restrict__ xb, const u16* __restrict__ Wt,
                           const float* __restrict__ bias, u16* __restrict__ out,
                           float scale) {
    const int w    = threadIdx.x >> 5;
    const int lane = threadIdx.x & 31;
    const int lh   = lane >> 4;
    const int nc   = lane & 15;
    const int gw   = blockIdx.x * 8 + w;
    const int NT   = Cn / 64;
    const int mbase = (gw / NT) * 16;
    const int nbase = (gw % NT) * 64;

    v8f acc[4] = {};
    gemm_16x64(xb + (size_t)(mbase + nc) * Cn, Wt + (size_t)nbase * Cn, lh, nc, acc);

    #pragma unroll
    for (int nt = 0; nt < 4; ++nt) {
        int n  = nbase + nt * 16 + nc;
        float bv = bias[n];
        int h = n / Dn, d = n % Dn;
        #pragma unroll
        for (int r = 0; r < 8; ++r) {
            int m = mbase + r + 8 * lh;               // C/D layout row
            int bb = m / Tn, t = m % Tn;
            float v = (acc[nt][r] + bv) * scale;
            out[(((size_t)bb * Hn + h) * Tn + t) * Dn + d] = f2bf(v);
        }
    }
}

// ---------------------------------------------------------------------------
// Output projection: d_out[m, n] = ctx @ Wo + bo, fp32 output.
// ---------------------------------------------------------------------------
__global__ void k_gemm_proj(const u16* __restrict__ ctxb, const u16* __restrict__ Wt,
                            const float* __restrict__ bias, float* __restrict__ out) {
    const int w    = threadIdx.x >> 5;
    const int lane = threadIdx.x & 31;
    const int lh   = lane >> 4;
    const int nc   = lane & 15;
    const int gw   = blockIdx.x * 8 + w;
    const int NT   = Cn / 64;
    const int mbase = (gw / NT) * 16;
    const int nbase = (gw % NT) * 64;

    v8f acc[4] = {};
    gemm_16x64(ctxb + (size_t)(mbase + nc) * Cn, Wt + (size_t)nbase * Cn, lh, nc, acc);

    #pragma unroll
    for (int nt = 0; nt < 4; ++nt) {
        int n = nbase + nt * 16 + nc;
        float bv = bias[n];
        #pragma unroll
        for (int r = 0; r < 8; ++r) {
            int m = mbase + r + 8 * lh;
            out[(size_t)m * Cn + n] = acc[nt][r] + bv;
        }
    }
}

// ---------------------------------------------------------------------------
// Flash attention: one wave per (b,h, 16-query tile). 32-key chunks.
//  - V chunk (contiguous 4 KB in [B,H,T,D]) staged LDS-side with
//    global_load_async_to_lds_b128, overlapped with the QK^T WMMAs + softmax.
//  - online softmax in C/D register layout; P re-shaped via LDS into A layout.
// ---------------------------------------------------------------------------
__global__ void k_attn(const u16* __restrict__ Q, const u16* __restrict__ K,
                       const u16* __restrict__ V, u16* __restrict__ ctx) {
    __shared__ __align__(16) u16 vbuf[8][32 * 64];   // per-wave V chunk (32 KiB)
    __shared__ __align__(16) u16 pbuf[8][16 * 32];   // per-wave P staging (8 KiB)

    const int w    = threadIdx.x >> 5;
    const int lane = threadIdx.x & 31;
    const int lh   = lane >> 4;
    const int nc   = lane & 15;
    const int gw   = blockIdx.x * 8 + w;
    const int tiles = Tn / 16;                       // 128
    const int t0   = (gw % tiles) * 16;
    const int bh   = gw / tiles;                     // 0..31
    const u16* Qp = Q + (size_t)bh * Tn * Dn;
    const u16* Kp = K + (size_t)bh * Tn * Dn;
    const u16* Vp = V + (size_t)bh * Tn * Dn;

    // Generic LDS pointer: low 32 bits == wave-relative LDS byte offset
    // (see ISA 10.2).
    const u32 vbuf_lds = (u32)(size_t)&vbuf[w][0];

    // Q tile in A layout, both D-slices (d0 = 0, 32). Q pre-scaled by 1/8.
    v16bf aq[2];
    const u16* qrow = Qp + (size_t)(t0 + nc) * Dn;
    #pragma unroll
    for (int kk = 0; kk < 2; ++kk) {
        u32x8 au;
        #pragma unroll
        for (int p = 0; p < 8; ++p)
            au[p] = *(const u32*)(qrow + kk * 32 + a_koff(p, lh));
        aq[kk] = __builtin_bit_cast(v16bf, au);
    }

    float mrow[8], lrow[8];
    #pragma unroll
    for (int r = 0; r < 8; ++r) { mrow[r] = -__builtin_inff(); lrow[r] = 0.f; }
    v8f O[4] = {};

    for (int s0 = 0; s0 <= t0 + 15; s0 += 32) {
        // ---- kick off async V chunk copy (4 KB contiguous -> LDS) ----
        // WAR guard: previous chunk's LDS gather (DScnt) must finish before
        // async writes (ASYNCcnt) land in the same buffer.
        asm volatile("s_wait_dscnt 0x0" ::: "memory");
        {
            u64 g = (u64)(Vp + (size_t)s0 * Dn);
            #pragma unroll
            for (int j = 0; j < 8; ++j) {
                u32 lo = vbuf_lds + (u32)(lane * 16 + j * 512);
                u64 ga = g + (u64)(lane * 16 + j * 512);
                asm volatile("global_load_async_to_lds_b128 %0, %1, off"
                             :: "v"(lo), "v"(ga) : "memory");
            }
        }

        // ---- S = Q @ K^T  (16 x 32 scores: two 16-col subtiles) ----
        v8f S[2] = {};
        #pragma unroll
        for (int kk = 0; kk < 2; ++kk) {
            #pragma unroll
            for (int nt = 0; nt < 2; ++nt) {
                // B lane = K-row (s0 + nt*16 + nc), contiguous along d
                const u16* krow = Kp + (size_t)(s0 + nt * 16 + nc) * Dn + kk * 32 + lh * 16;
                u32x8 bu;
                #pragma unroll
                for (int p = 0; p < 8; ++p) bu[p] = *(const u32*)(krow + 2 * p);
                v16bf bk = __builtin_bit_cast(v16bf, bu);
                S[nt] = __builtin_amdgcn_wmma_f32_16x16x32_bf16(
                    false, aq[kk], false, bk, (short)0, S[nt], false, false);
            }
        }

        // ---- causal mask + online softmax (data-level, EXEC stays full) ----
        #pragma unroll
        for (int r = 0; r < 8; ++r) {
            int t = t0 + r + 8 * lh;
            float sv0 = (s0 + nc      > t) ? -__builtin_inff() : S[0][r];
            float sv1 = (s0 + 16 + nc > t) ? -__builtin_inff() : S[1][r];
            float cm  = half16_max(fmaxf(sv0, sv1));
            float nm  = fmaxf(mrow[r], cm);
            float corr = __expf(mrow[r] - nm);
            float p0 = __expf(sv0 - nm);
            float p1 = __expf(sv1 - nm);
            lrow[r] = lrow[r] * corr + half16_sum(p0 + p1);
            mrow[r] = nm;
            #pragma unroll
            for (int nt = 0; nt < 4; ++nt) O[nt][r] *= corr;
            int row = r + 8 * lh;
            pbuf[w][row * 32 + nc]      = f2bf(p0);
            pbuf[w][row * 32 + 16 + nc] = f2bf(p1);
        }

        // ---- reload P in A-operand layout (same-wave DS ops are in-order) ----
        u32x8 pu;
        #pragma unroll
        for (int p = 0; p < 8; ++p)
            pu[p] = *(const u32*)&pbuf[w][nc * 32 + a_koff(p, lh)];
        v16bf pa = __builtin_bit_cast(v16bf, pu);

        // ---- wait for async V, gather B operand from LDS, O += P @ V ----
        asm volatile("s_wait_asynccnt 0x0" ::: "memory");
        #pragma unroll
        for (int nt = 0; nt < 4; ++nt) {
            int dcol = nt * 16 + nc;
            u16x16 vu;
            #pragma unroll
            for (int e = 0; e < 16; ++e)
                vu[e] = vbuf[w][(lh * 16 + e) * Dn + dcol];
            v16bf bv = __builtin_bit_cast(v16bf, vu);
            O[nt] = __builtin_amdgcn_wmma_f32_16x16x32_bf16(
                false, pa, false, bv, (short)0, O[nt], false, false);
        }
    }

    // ---- normalize and write context in [B,T,H,D] for the out projection ----
    int b = bh / Hn, h = bh % Hn;
    #pragma unroll
    for (int nt = 0; nt < 4; ++nt) {
        int d = nt * 16 + nc;
        #pragma unroll
        for (int r = 0; r < 8; ++r) {
            int t = t0 + r + 8 * lh;
            float v = O[nt][r] / lrow[r];
            ctx[(((size_t)b * Tn + t) * Hn + h) * Dn + d] = f2bf(v);
        }
    }
}

// ---------------------------------------------------------------------------
// Host launch
// ---------------------------------------------------------------------------
extern "C" void kernel_launch(void* const* d_in, const int* in_sizes, int n_in,
                              void* d_out, int out_size, void* d_ws, size_t ws_size,
                              hipStream_t stream) {
    const float* x  = (const float*)d_in[0];
    const float* Wq = (const float*)d_in[1];
    const float* bq = (const float*)d_in[2];
    const float* Wk = (const float*)d_in[3];
    const float* bk = (const float*)d_in[4];
    const float* Wv = (const float*)d_in[5];
    const float* bv = (const float*)d_in[6];
    const float* Wo = (const float*)d_in[7];
    const float* bo = (const float*)d_in[8];

    char* ws = (char*)d_ws;
    // workspace layout (bytes); total = 48 MiB
    u16* xb   = (u16*)(ws);                                  //  8 MiB: x bf16
    u16* Wqt  = (u16*)(ws + (size_t)8  * 1024 * 1024);       //  2 MiB each, transposed
    u16* Wkt  = (u16*)(ws + (size_t)10 * 1024 * 1024);
    u16* Wvt  = (u16*)(ws + (size_t)12 * 1024 * 1024);
    u16* Wot  = (u16*)(ws + (size_t)14 * 1024 * 1024);
    u16* Qb   = (u16*)(ws + (size_t)16 * 1024 * 1024);       //  8 MiB [B,H,T,D]
    u16* Kb   = (u16*)(ws + (size_t)24 * 1024 * 1024);
    u16* Vb   = (u16*)(ws + (size_t)32 * 1024 * 1024);
    u16* ctx  = (u16*)(ws + (size_t)40 * 1024 * 1024);       //  8 MiB [B,T,H,D]

    const int nX = Mn * Cn;          // 4 Mi elements
    const int nW = Cn * Cn;          // 1 Mi elements

    k_f32_to_bf16<<<nX / 256, 256, 0, stream>>>(x, xb, nX);
    k_w_transpose_bf16<<<nW / 256, 256, 0, stream>>>(Wq, Wqt);
    k_w_transpose_bf16<<<nW / 256, 256, 0, stream>>>(Wk, Wkt);
    k_w_transpose_bf16<<<nW / 256, 256, 0, stream>>>(Wv, Wvt);
    k_w_transpose_bf16<<<nW / 256, 256, 0, stream>>>(Wo, Wot);

    // 4096 waves per GEMM = 512 blocks x 8 waves
    k_gemm_qkv<<<512, 256, 0, stream>>>(xb, Wqt, bq, Qb, 0.125f);  // 1/sqrt(64)
    k_gemm_qkv<<<512, 256, 0, stream>>>(xb, Wkt, bk, Kb, 1.0f);
    k_gemm_qkv<<<512, 256, 0, stream>>>(xb, Wvt, bv, Vb, 1.0f);

    // 4096 waves: one per (b,h, 16-query tile)
    k_attn<<<512, 256, 0, stream>>>(Qb, Kb, Vb, ctx);

    k_gemm_proj<<<512, 256, 0, stream>>>(ctx, Wot, bo, (float*)d_out);
}
